// BitNetDiTTriton_77421080477792
// MI455X (gfx1250) — 8-bit matrix operand order repaired and checked by reading, compiled, not run on hardware
//
#include <hip/hip_runtime.h>
#include <hip/hip_bf16.h>
#include <math.h>

typedef __attribute__((ext_vector_type(8)))  int      v8i;
typedef __attribute__((ext_vector_type(8)))  float    v8f;
typedef __attribute__((ext_vector_type(16))) _Float16 v16h;

constexpr int kDim   = 768;
constexpr int kHeads = 12;
constexpr int kHd    = 64;
constexpr int kTok   = 256;
constexpr int kBatch = 16;
constexpr int kHid   = 3072;
constexpr int kQkvN  = 2304;
constexpr int kM     = kBatch * kTok;

__device__ __forceinline__ float gelu_f(float x) {
  const float c0 = 0.7978845608028654f;
  return 0.5f * x * (1.f + tanhf(c0 * (x + 0.044715f * x * x * x)));
}
__device__ __forceinline__ float silu_f(float x) { return x / (1.f + __expf(-x)); }

template <typename F>
__device__ __forceinline__ v16h make_frag16(int lane, F at) {
  v16h f;
  const int kh = (lane >> 4) << 3;
#pragma unroll
  for (int v = 0; v < 8; ++v) {
    const int kk = kh + 2 * (v & 3) + ((v & 4) << 2);
    f[2 * v]     = (_Float16)at(kk);
    f[2 * v + 1] = (_Float16)at(kk + 1);
  }
  return f;
}

__device__ __forceinline__ v8i load_a8(const signed char* p) {
  int2 t0 = *reinterpret_cast<const int2*>(p);
  int2 t1 = *reinterpret_cast<const int2*>(p + 16);
  int2 t2 = *reinterpret_cast<const int2*>(p + 32);
  int2 t3 = *reinterpret_cast<const int2*>(p + 48);
  v8i f;
  f[0] = t0.x; f[1] = t0.y; f[2] = t1.x; f[3] = t1.y;
  f[4] = t2.x; f[5] = t2.y; f[6] = t3.x; f[7] = t3.y;
  return f;
}
__device__ __forceinline__ v8i load_b8(const signed char* p) {
  int4 u0 = *reinterpret_cast<const int4*>(p);
  int4 u1 = *reinterpret_cast<const int4*>(p + 32);
  v8i f;
  f[0] = u0.x; f[1] = u0.y; f[2] = u0.z; f[3] = u0.w;
  f[4] = u1.x; f[5] = u1.y; f[6] = u1.z; f[7] = u1.w;
  return f;
}

__global__ void __launch_bounds__(256)
absmean_partial(const float* __restrict__ w, float* __restrict__ part, int n) {
  __shared__ float red[256];
  float acc = 0.f;
  for (int i = blockIdx.x * 256 + threadIdx.x; i < n; i += gridDim.x * 256)
    acc += fabsf(w[i]);
  red[threadIdx.x] = acc; __syncthreads();
  for (int s = 128; s > 0; s >>= 1) {
    if (threadIdx.x < s) red[threadIdx.x] += red[threadIdx.x + s];
    __syncthreads();
  }
  if (threadIdx.x == 0) part[blockIdx.x] = red[0];
}

__global__ void __launch_bounds__(512)
finalize_scale(const float* __restrict__ part, int nparts, int numel,
               float* __restrict__ scale) {
  __shared__ float red[512];
  red[threadIdx.x] = (threadIdx.x < nparts) ? part[threadIdx.x] : 0.f;
  __syncthreads();
  for (int s = 256; s > 0; s >>= 1) {
    if (threadIdx.x < s) red[threadIdx.x] += red[threadIdx.x + s];
    __syncthreads();
  }
  if (threadIdx.x == 0) scale[0] = red[0] / (float)numel + 1e-5f;
}

__global__ void __launch_bounds__(256)
quantize_weight(const float* __restrict__ w, const float* __restrict__ scale,
                signed char* __restrict__ wq, int n) {
  int i = blockIdx.x * 256 + threadIdx.x;
  if (i >= n) return;
  float r = rintf(w[i] / scale[0]);
  r = fminf(1.f, fmaxf(-1.f, r));
  wq[i] = (signed char)r;
}

__global__ void __launch_bounds__(256)
norm_mod_quant(const float* __restrict__ x, const float* __restrict__ nw,
               const float* __restrict__ mod, int shift_off, int scale_off,
               signed char* __restrict__ q, float* __restrict__ ainv) {
  __shared__ float red[256];
  const int row = blockIdx.x;
  const int tid = threadIdx.x;
  const int bb  = row >> 8;
  const float* xr = x + (size_t)row * kDim;
  const float* sh = mod + (size_t)bb * (4 * kDim) + shift_off;
  const float* sc = mod + (size_t)bb * (4 * kDim) + scale_off;

  float v[3], ssq = 0.f;
#pragma unroll
  for (int j = 0; j < 3; ++j) { float t = xr[tid + 256 * j]; v[j] = t; ssq += t * t; }
  red[tid] = ssq; __syncthreads();
  for (int s = 128; s > 0; s >>= 1) { if (tid < s) red[tid] += red[tid + s]; __syncthreads(); }
  const float r = rsqrtf(red[0] / (float)kDim + 1e-6f);
  __syncthreads();

  float y[3], mx = 0.f;
#pragma unroll
  for (int j = 0; j < 3; ++j) {
    const int col = tid + 256 * j;
    float tv = v[j] * r * nw[col] * (1.f + sc[col]) + sh[col];
    y[j] = tv; mx = fmaxf(mx, fabsf(tv));
  }
  red[tid] = mx; __syncthreads();
  for (int s = 128; s > 0; s >>= 1) { if (tid < s) red[tid] = fmaxf(red[tid], red[tid + s]); __syncthreads(); }
  const float amax = fmaxf(red[0], 1e-5f);
  const float xs = 127.f / amax;
#pragma unroll
  for (int j = 0; j < 3; ++j) {
    const int col = tid + 256 * j;
    float qv = fminf(127.f, fmaxf(-128.f, rintf(y[j] * xs)));
    q[(size_t)row * kDim + col] = (signed char)qv;
  }
  if (tid == 0) ainv[row] = amax / 127.f;
}

__global__ void __launch_bounds__(256)
quant_rows(const float* __restrict__ x, signed char* __restrict__ q,
           float* __restrict__ ainv, int K) {
  __shared__ float red[256];
  const int row = blockIdx.x;
  const int tid = threadIdx.x;
  const float* xr = x + (size_t)row * K;
  const int per = K >> 8;
  float v[12];
  float mx = 0.f;
  for (int j = 0; j < per; ++j) {
    float t = xr[tid + 256 * j]; v[j] = t; mx = fmaxf(mx, fabsf(t));
  }
  red[tid] = mx; __syncthreads();
  for (int s = 128; s > 0; s >>= 1) { if (tid < s) red[tid] = fmaxf(red[tid], red[tid + s]); __syncthreads(); }
  const float amax = fmaxf(red[0], 1e-5f);
  const float xs = 127.f / amax;
  for (int j = 0; j < per; ++j) {
    float qv = fminf(127.f, fmaxf(-128.f, rintf(v[j] * xs)));
    q[(size_t)row * K + tid + 256 * j] = (signed char)qv;
  }
  if (tid == 0) ainv[row] = amax / 127.f;
}

__global__ void __launch_bounds__(256)
rmsnorm_final(const float* __restrict__ x, const float* __restrict__ w,
              float* __restrict__ y) {
  __shared__ float red[256];
  const int row = blockIdx.x;
  const int tid = threadIdx.x;
  const float* xr = x + (size_t)row * kDim;
  float v[3], ssq = 0.f;
#pragma unroll
  for (int j = 0; j < 3; ++j) { float t = xr[tid + 256 * j]; v[j] = t; ssq += t * t; }
  red[tid] = ssq; __syncthreads();
  for (int s = 128; s > 0; s >>= 1) { if (tid < s) red[tid] += red[tid + s]; __syncthreads(); }
  const float r = rsqrtf(red[0] / (float)kDim + 1e-6f);
#pragma unroll
  for (int j = 0; j < 3; ++j) {
    const int col = tid + 256 * j;
    y[(size_t)row * kDim + col] = v[j] * r * w[col];
  }
}

__global__ void __launch_bounds__(256)
gemm_i8_wmma(const signed char* __restrict__ A, const signed char* __restrict__ W,
             const float* __restrict__ a_inv, const float* __restrict__ w_scale,
             const float* res, float* out, int N, int K, int ep) {
  const int lane = threadIdx.x & 31;
  const int wave = threadIdx.x >> 5;
  const int l15  = lane & 15;
  const int m0 = blockIdx.y << 4;
  const int n0 = (blockIdx.x << 8) + (wave << 5);

  const signed char* pa  = A + (size_t)(m0 + l15) * K + ((lane >> 4) << 4);
  const signed char* pb0 = W + (size_t)(n0 + l15) * K + ((lane >> 4) << 4);
  const signed char* pb1 = W + (size_t)(n0 + 16 + l15) * K + ((lane >> 4) << 4);

  v8i a  = load_b8(pa);
  v8i b0 = load_b8(pb0);
  v8i b1 = load_b8(pb1);
  v8i acc0 = {0, 0, 0, 0, 0, 0, 0, 0};
  v8i acc1 = {0, 0, 0, 0, 0, 0, 0, 0};

  for (int k0 = 0; k0 < K - 64; k0 += 64) {
    v8i an  = load_b8(pa + k0 + 64);
    v8i b0n = load_b8(pb0 + k0 + 64);
    v8i b1n = load_b8(pb1 + k0 + 64);
    __builtin_prefetch(pb0 + k0 + 128, 0, 1);
    __builtin_prefetch(pb1 + k0 + 128, 0, 1);
    acc0 = __builtin_amdgcn_wmma_i32_16x16x64_iu8(true, a, true, b0, acc0, false, false);
    acc1 = __builtin_amdgcn_wmma_i32_16x16x64_iu8(true, a, true, b1, acc1, false, false);
    a = an; b0 = b0n; b1 = b1n;
  }
  acc0 = __builtin_amdgcn_wmma_i32_16x16x64_iu8(true, a, true, b0, acc0, false, false);
  acc1 = __builtin_amdgcn_wmma_i32_16x16x64_iu8(true, a, true, b1, acc1, false, false);

  const float ws = w_scale[0];
  const int mb = m0 + ((lane >> 4) << 3);
#pragma unroll
  for (int v = 0; v < 8; ++v) {
    const int m = mb + v;
    const float dq = ws * a_inv[m];
    {
      const size_t idx = (size_t)m * N + (n0 + l15);
      float val = (float)acc0[v] * dq;
      if (ep == 1)      val += res[idx];
      else if (ep == 2) val = gelu_f(val);
      out[idx] = val;
    }
    {
      const size_t idx = (size_t)m * N + (n0 + 16 + l15);
      float val = (float)acc1[v] * dq;
      if (ep == 1)      val += res[idx];
      else if (ep == 2) val = gelu_f(val);
      out[idx] = val;
    }
  }
}

__global__ void __launch_bounds__(32)
attn_wmma(const float* __restrict__ qkv, float* __restrict__ o) {
  const int lane = threadIdx.x;
  const int rb = blockIdx.x;
  const int h  = blockIdx.y;
  const int b  = blockIdx.z;
  const int l15 = lane & 15;

  __shared__ float    ss[16][kTok];
  __shared__ _Float16 sp[16][kTok + 16];

  const size_t qbase = ((size_t)(b * kTok + (rb << 4) + l15)) * kQkvN + h * kHd;
  v16h qa0 = make_frag16(lane, [&](int kk) { return qkv[qbase + kk]; });
  v16h qa1 = make_frag16(lane, [&](int kk) { return qkv[qbase + 32 + kk]; });

  for (int ct = 0; ct < 16; ++ct) {
    const size_t kb = ((size_t)(b * kTok + (ct << 4) + l15)) * kQkvN + kDim + h * kHd;
    v16h k0 = make_frag16(lane, [&](int kk) { return qkv[kb + kk]; });
    v16h k1 = make_frag16(lane, [&](int kk) { return qkv[kb + 32 + kk]; });
    v8f acc = {0.f, 0.f, 0.f, 0.f, 0.f, 0.f, 0.f, 0.f};
    acc = __builtin_amdgcn_wmma_f32_16x16x32_f16(false, qa0, false, k0, (short)0, acc, false, false);
    acc = __builtin_amdgcn_wmma_f32_16x16x32_f16(false, qa1, false, k1, (short)0, acc, false, false);
    const int mb = (lane >> 4) << 3;
#pragma unroll
    for (int v = 0; v < 8; ++v) ss[mb + v][(ct << 4) + l15] = acc[v] * 0.125f;
  }
  __syncthreads();

  {
    const int r   = l15;
    const int off = (lane >> 4) * 128;
    float mx = -1e30f;
    for (int j = 0; j < 128; ++j) mx = fmaxf(mx, ss[r][off + j]);
    mx = fmaxf(mx, __shfl_xor(mx, 16, 32));
    float sum = 0.f;
    for (int j = 0; j < 128; ++j) {
      float e = __expf(ss[r][off + j] - mx);
      ss[r][off + j] = e;
      sum += e;
    }
    sum += __shfl_xor(sum, 16, 32);
    const float inv = 1.f / sum;
    for (int j = 0; j < 128; ++j) sp[r][off + j] = (_Float16)(ss[r][off + j] * inv);
  }
  __syncthreads();

  for (int dt = 0; dt < 4; ++dt) {
    const int d = (dt << 4) + l15;
    v8f acc = {0.f, 0.f, 0.f, 0.f, 0.f, 0.f, 0.f, 0.f};
    for (int kc = 0; kc < 8; ++kc) {
      v16h pa = make_frag16(lane, [&](int kk) { return sp[l15][(kc << 5) + kk]; });
      v16h vb = make_frag16(lane, [&](int kk) {
        return qkv[((size_t)(b * kTok + (kc << 5) + kk)) * kQkvN + 2 * kDim + h * kHd + d];
      });
      acc = __builtin_amdgcn_wmma_f32_16x16x32_f16(false, pa, false, vb, (short)0, acc, false, false);
    }
    const int mb = (lane >> 4) << 3;
#pragma unroll
    for (int v = 0; v < 8; ++v)
      o[((size_t)(b * kTok + (rb << 4) + mb + v)) * kDim + h * kHd + (dt << 4) + l15] = acc[v];
  }
}

__global__ void __launch_bounds__(256)
patch_embed_k(const float* __restrict__ x, const float* __restrict__ pw,
              const float* __restrict__ pb, const float* __restrict__ pos,
              float* __restrict__ z) {
  const int o = blockIdx.x * 256 + threadIdx.x;
  const int m = blockIdx.y;
  const int b = m >> 8, tokn = m & 255;
  const int hi = tokn >> 4, wi = tokn & 15;
  const float* wr = pw + (size_t)o * 768;
  float acc = 0.f;
  for (int k = 0; k < 768; ++k) {
    const int c = k >> 8, rem = k & 255;
    const int py = rem >> 4, px = rem & 15;
    const float xv = x[(((size_t)(b * 3 + c) * 256) + hi * 16 + py) * 256 + wi * 16 + px];
    acc = fmaf(xv, wr[k], acc);
  }
  z[(size_t)m * 768 + o] = acc + pb[o] + pos[(size_t)tokn * 768 + o];
}

__global__ void __launch_bounds__(256)
time_emb_k(const float* __restrict__ t, float* __restrict__ emb) {
  int idx = blockIdx.x * 256 + threadIdx.x;
  if (idx >= kBatch * kDim) return;
  int i = idx / kDim, j = idx % kDim;
  const int half = kDim / 2;
  int jj = (j < half) ? j : j - half;
  float freq = __expf(-logf(10000.f) * (float)jj / (float)(half - 1));
  float a = t[i] * freq;
  emb[idx] = (j < half) ? sinf(a) : cosf(a);
}

__global__ void __launch_bounds__(256)
dense_gemm(const float* __restrict__ X, const float* __restrict__ W,
           const float* __restrict__ bias, float* __restrict__ Y,
           int N, int K, int act_in, int act_out) {
  const int n = blockIdx.x * 256 + threadIdx.x;
  const int m = blockIdx.y;
  if (n >= N) return;
  const float* xr = X + (size_t)m * K;
  const float* wr = W + (size_t)n * K;
  float acc = 0.f;
  for (int k = 0; k < K; ++k) {
    float xv = xr[k];
    if (act_in) xv = silu_f(xv);
    acc = fmaf(xv, wr[k], acc);
  }
  acc += bias[n];
  if (act_out) acc = gelu_f(acc);
  Y[(size_t)m * N + n] = acc;
}

__global__ void __launch_bounds__(256)
unpatch_k(const float* __restrict__ zh, float* __restrict__ out) {
  int i = blockIdx.x * 256 + threadIdx.x;
  const int total = kBatch * 3 * 256 * 256;
  if (i >= total) return;
  const int ww = i & 255;
  const int hh = (i >> 8) & 255;
  const int tcb = i >> 16;
  const int c = tcb % 3, b = tcb / 3;
  const int hi = hh >> 4, py = hh & 15;
  const int wi = ww >> 4, px = ww & 15;
  const int tok = hi * 16 + wi;
  const int k = c * 256 + py * 16 + px;
  out[i] = zh[((size_t)(b * 256 + tok)) * 768 + k];
}

extern "C" void kernel_launch(void* const* d_in, const int* in_sizes, int n_in,
                              void* d_out, int out_size, void* d_ws, size_t ws_size,
                              hipStream_t stream) {
  (void)in_sizes; (void)n_in; (void)out_size; (void)ws_size;

  const float* x       = (const float*)d_in[0];
  const float* t       = (const float*)d_in[1];
  const float* patch_w = (const float*)d_in[2];
  const float* patch_b = (const float*)d_in[3];
  const float* pos     = (const float*)d_in[4];
  const float* t_w1    = (const float*)d_in[5];
  const float* t_b1    = (const float*)d_in[6];
  const float* t_w2    = (const float*)d_in[7];
  const float* t_b2    = (const float*)d_in[8];
  const float* bn1     = (const float*)d_in[9];
  const float* bqkv    = (const float*)d_in[10];
  const float* bproj   = (const float*)d_in[11];
  const float* bn2     = (const float*)d_in[12];
  const float* bfc1    = (const float*)d_in[13];
  const float* bfc2    = (const float*)d_in[14];
  const float* badaw   = (const float*)d_in[15];
  const float* badab   = (const float*)d_in[16];
  const float* normw   = (const float*)d_in[17];
  const float* headw   = (const float*)d_in[18];
  const float* headb   = (const float*)d_in[19];

  char* wsp = (char*)d_ws;
  auto carve = [&](size_t bytes) -> void* {
    void* p = (void*)wsp;
    wsp += (bytes + 255) & ~(size_t)255;
    return p;
  };
  float*       z     = (float*)carve((size_t)kM * kDim * 4);
  float*       emb   = (float*)carve((size_t)kBatch * kDim * 4);
  float*       th    = (float*)carve((size_t)kBatch * kDim * 4);
  float*       temb  = (float*)carve((size_t)kBatch * kDim * 4);
  float*       mod   = (float*)carve((size_t)kBatch * 4 * kDim * 4);
  signed char* aq    = (signed char*)carve((size_t)kM * kHid);
  float*       ainv  = (float*)carve((size_t)kM * 4);
  signed char* wq    = (signed char*)carve((size_t)kHid * kDim);
  float*       part  = (float*)carve(512 * 4);
  float*       wsc   = (float*)carve(64);
  float*       gout  = (float*)carve((size_t)kM * kHid * 4);
  float*       attnb = (float*)carve((size_t)kM * kDim * 4);

  auto quant_weight = [&](const float* wmat, int numel) {
    absmean_partial<<<512, 256, 0, stream>>>(wmat, part, numel);
    finalize_scale<<<1, 512, 0, stream>>>(part, 512, numel, wsc);
    quantize_weight<<<(numel + 255) / 256, 256, 0, stream>>>(wmat, wsc, wq, numel);
  };
  auto gemm = [&](int N, int K, const float* res, float* out, int ep) {
    gemm_i8_wmma<<<dim3(N / 256, kM / 16), 256, 0, stream>>>(aq, wq, ainv, wsc, res, out, N, K, ep);
  };

  patch_embed_k<<<dim3(3, kM), 256, 0, stream>>>(x, patch_w, patch_b, pos, z);

  time_emb_k<<<(kBatch * kDim + 255) / 256, 256, 0, stream>>>(t, emb);
  dense_gemm<<<dim3(3, kBatch), 256, 0, stream>>>(emb, t_w1, t_b1, th, kDim, kDim, 0, 1);
  dense_gemm<<<dim3(3, kBatch), 256, 0, stream>>>(th, t_w2, t_b2, temb, kDim, kDim, 0, 0);

  for (int i = 0; i < 12; ++i) {
    const float* qkv_w = bqkv  + (size_t)i * kQkvN * kDim;
    const float* prj_w = bproj + (size_t)i * kDim * kDim;
    const float* fc1_w = bfc1  + (size_t)i * kHid * kDim;
    const float* fc2_w = bfc2  + (size_t)i * kDim * kHid;
    const float* ada_w = badaw + (size_t)i * 4 * kDim * kDim;
    const float* ada_b = badab + (size_t)i * 4 * kDim;
    const float* n1    = bn1 + (size_t)i * kDim;
    const float* n2    = bn2 + (size_t)i * kDim;

    dense_gemm<<<dim3(12, kBatch), 256, 0, stream>>>(temb, ada_w, ada_b, mod, 4 * kDim, kDim, 1, 0);

    quant_weight(qkv_w, kQkvN * kDim);
    norm_mod_quant<<<kM, 256, 0, stream>>>(z, n1, mod, 0, kDim, aq, ainv);
    gemm(kQkvN, kDim, nullptr, gout, 0);

    attn_wmma<<<dim3(16, kHeads, kBatch), 32, 0, stream>>>(gout, attnb);

    quant_weight(prj_w, kDim * kDim);
    quant_rows<<<kM, 256, 0, stream>>>(attnb, aq, ainv, kDim);
    gemm(kDim, kDim, z, z, 1);

    quant_weight(fc1_w, kHid * kDim);
    norm_mod_quant<<<kM, 256, 0, stream>>>(z, n2, mod, 2 * kDim, 3 * kDim, aq, ainv);
    gemm(kHid, kDim, nullptr, gout, 2);

    quant_weight(fc2_w, kDim * kHid);
    quant_rows<<<kM, 256, 0, stream>>>(gout, aq, ainv, kHid);
    gemm(kDim, kHid, z, z, 1);
  }

  rmsnorm_final<<<kM, 256, 0, stream>>>(z, normw, attnb);
  dense_gemm<<<dim3(3, kM), 256, 0, stream>>>(attnb, headw, headb, gout, kDim, kDim, 0, 0);
  unpatch_k<<<(kBatch * 3 * 256 * 256 + 255) / 256, 256, 0, stream>>>(gout, (float*)d_out);
}
